// GraphAttentionNetwork_41412074668700
// MI455X (gfx1250) — matrix operand order repaired and checked by reading, not compiled, not run on hardware
//
#include <hip/hip_runtime.h>
#include <cmath>

// ---------------------------------------------------------------------------
// CDNA5 (gfx1250) GAT forward: 2x { bf16-WMMA GEMM -> attention softmax over
// incoming edges -> atomic scatter -> fused LayerNorm+ELU+residual }.
// GEMM: one wave computes a 16x64 output strip (4 WMMA tiles), A-fragment
// reused across the 4 N-tiles -> 2.5 loads per v_wmma_f32_16x16x32_bf16.
// ---------------------------------------------------------------------------

typedef __attribute__((ext_vector_type(16))) __bf16 v16bf;
typedef __attribute__((ext_vector_type(8)))  float  v8f;

union BFrag {
  v16bf v;
  uint4 q[2];
};

__device__ __forceinline__ unsigned short f2bf(float f) {
  unsigned int u = __float_as_uint(f);
  unsigned int r = (u + 0x7FFFu + ((u >> 16) & 1u)) >> 16;  // RNE
  return (unsigned short)r;
}

// float atomic max via sign-split integer atomics (init value must be -inf)
__device__ __forceinline__ void atomicMaxF(float* addr, float val) {
  if (val >= 0.0f) {
    atomicMax((int*)addr, __float_as_int(val));
  } else {
    atomicMin((unsigned int*)addr, __float_as_uint(val));
  }
}

// --------------------------- conversion kernels ----------------------------

__global__ void cvt_bf16_kernel(const float* __restrict__ src,
                                unsigned short* __restrict__ dst, size_t n) {
  size_t i = (size_t)blockIdx.x * blockDim.x + threadIdx.x;
  if (i < n) dst[i] = f2bf(src[i]);
}

// W (512x512 row-major, f32) -> WT (512x512, WT[j][k] = W[k][j], bf16)
__global__ void cvt_w_transpose_kernel(const float* __restrict__ W,
                                       unsigned short* __restrict__ WT) {
  int i = blockIdx.x * blockDim.x + threadIdx.x;  // 262144 threads
  if (i < 512 * 512) {
    int k = i >> 9;
    int j = i & 511;
    WT[(size_t)j * 512 + k] = f2bf(W[i]);
  }
}

// ------------------------------ WMMA GEMM ----------------------------------
// h[M x 512] = xb[M x 512] @ W[512 x 512]   (bf16 in, f32 accumulate)
// One wave computes a 16x64 strip: A fragment loaded once per k-step and
// reused across 4 B fragments / 4 WMMAs. Fragments use contiguous K order (lane half h holds K=16h..16h+15) for A and B.

__global__ void gemm_bf16_wmma_kernel(const unsigned short* __restrict__ xb,
                                      const unsigned short* __restrict__ wt,
                                      float* __restrict__ h, int total_waves) {
  const int wave = (int)((blockIdx.x * blockDim.x + threadIdx.x) >> 5);
  if (wave >= total_waves) return;          // wave-uniform: EXEC all-ones inside
  const int lane = threadIdx.x & 31;
  const int tm = wave >> 3;                 // 16-row tile index
  const int tg = wave & 7;                  // 64-col group index (512/64 = 8)

  const int m       = lane & 15;
  const int row     = tm * 16 + m;          // A row
  const int colBase = tg * 64 + (lane & 15);
  const int koffA   = (lane < 16) ? 0 : 16; // A: lanes 16-31 hold K=16..31
  const int koffB   = (lane < 16) ? 0 : 16; // B: lanes 16-31 hold K=16..31

  const unsigned short* arow = xb + (size_t)row * 512;
  const unsigned short* bp0  = wt + (size_t)(colBase +  0) * 512 + koffB;
  const unsigned short* bp1  = wt + (size_t)(colBase + 16) * 512 + koffB;
  const unsigned short* bp2  = wt + (size_t)(colBase + 32) * 512 + koffB;
  const unsigned short* bp3  = wt + (size_t)(colBase + 48) * 512 + koffB;

  v8f acc0 = {}, acc1 = {}, acc2 = {}, acc3 = {};
#pragma unroll 2
  for (int k = 0; k < 512; k += 32) {
    BFrag a, b0, b1, b2, b3;
    // A: a[i] = x[row][k+koffA+i], i = 0..15 (contiguous along K)
    a.q[0] = *(const uint4*)(arow + k + koffA);
    a.q[1] = *(const uint4*)(arow + k + koffA + 8);
    // B: b[i] = WT[col][k+koffB+i], i = 0..15 (contiguous along K)
    b0.q[0] = *(const uint4*)(bp0 + k);
    b0.q[1] = *(const uint4*)(bp0 + k + 8);
    b1.q[0] = *(const uint4*)(bp1 + k);
    b1.q[1] = *(const uint4*)(bp1 + k + 8);
    b2.q[0] = *(const uint4*)(bp2 + k);
    b2.q[1] = *(const uint4*)(bp2 + k + 8);
    b3.q[0] = *(const uint4*)(bp3 + k);
    b3.q[1] = *(const uint4*)(bp3 + k + 8);
    acc0 = __builtin_amdgcn_wmma_f32_16x16x32_bf16(false, a.v, false, b0.v,
                                                   (short)0, acc0, false, false);
    acc1 = __builtin_amdgcn_wmma_f32_16x16x32_bf16(false, a.v, false, b1.v,
                                                   (short)0, acc1, false, false);
    acc2 = __builtin_amdgcn_wmma_f32_16x16x32_bf16(false, a.v, false, b2.v,
                                                   (short)0, acc2, false, false);
    acc3 = __builtin_amdgcn_wmma_f32_16x16x32_bf16(false, a.v, false, b3.v,
                                                   (short)0, acc3, false, false);
  }

  // D layout: VGPR r -> row (r or r+8), col = lane&15
  const int rbase = (lane < 16) ? 0 : 8;
  float* hrow = h + (size_t)(tm * 16 + rbase) * 512 + tg * 64 + (lane & 15);
#pragma unroll
  for (int r = 0; r < 8; ++r) {
    float* hp = hrow + (size_t)r * 512;
    hp[0]  = acc0[r];
    hp[16] = acc1[r];
    hp[32] = acc2[r];
    hp[48] = acc3[r];
  }
}

// ------------------------- attention / softmax -----------------------------

// alpha_s[n,hd] = sum_c h[n,hd*C+c]*a_s[hd,c]   (same for alpha_d)
__global__ void alpha_kernel(const float* __restrict__ h,
                             const float* __restrict__ a_s,
                             const float* __restrict__ a_d,
                             float* __restrict__ as_out, float* __restrict__ ad_out,
                             int N, int H, int C) {
  int i = blockIdx.x * blockDim.x + threadIdx.x;  // n*H + hd
  if (i >= N * H) return;
  int n = i / H, hd = i % H;
  const float* hp  = h + (size_t)n * 512 + hd * C;
  const float* asp = a_s + hd * C;
  const float* adp = a_d + hd * C;
  float ss = 0.0f, sd = 0.0f;
  for (int c = 0; c < C; ++c) {
    float v = hp[c];
    ss += v * asp[c];
    sd += v * adp[c];
  }
  as_out[i] = ss;
  ad_out[i] = sd;
}

__global__ void fill_kernel(float* __restrict__ p, float v, size_t n) {
  size_t i = (size_t)blockIdx.x * blockDim.x + threadIdx.x;
  size_t stride = (size_t)gridDim.x * blockDim.x;
  for (; i < n; i += stride) p[i] = v;
}

// logits[e,hd] = leaky_relu(as[src]+ad[dst]); atomic segment max over dst
__global__ void edge_logits_kernel(const int* __restrict__ ei, int E, int N, int H,
                                   const float* __restrict__ as,
                                   const float* __restrict__ ad,
                                   float* __restrict__ logits,
                                   float* __restrict__ seg_max) {
  long long idx = (long long)blockIdx.x * blockDim.x + threadIdx.x;
  long long tot = (long long)(E + N) * H;
  if (idx >= tot) return;
  int e = (int)(idx / H), hd = (int)(idx % H);
  int s, d;
  if (e < E) { s = ei[e]; d = ei[E + e]; } else { s = d = e - E; }  // self-loops
  float l = as[(size_t)s * H + hd] + ad[(size_t)d * H + hd];
  l = (l > 0.0f) ? l : 0.2f * l;  // leaky relu, neg slope 0.2
  logits[idx] = l;
  atomicMaxF(&seg_max[(size_t)d * H + hd], l);
}

// logits <- exp(logits - seg_max[dst]); atomic segment sum over dst
__global__ void edge_exp_kernel(const int* __restrict__ ei, int E, int N, int H,
                                float* __restrict__ logits,
                                const float* __restrict__ seg_max,
                                float* __restrict__ seg_sum) {
  long long idx = (long long)blockIdx.x * blockDim.x + threadIdx.x;
  long long tot = (long long)(E + N) * H;
  if (idx >= tot) return;
  int e = (int)(idx / H), hd = (int)(idx % H);
  int d = (e < E) ? ei[E + e] : (e - E);
  float v = expf(logits[idx] - seg_max[(size_t)d * H + hd]);
  logits[idx] = v;
  atomicAdd(&seg_sum[(size_t)d * H + hd], v);
}

// acc[dst, :] += h[src, :] * (e / seg_sum[dst]) — one 128-thread block per edge
__global__ void scatter_kernel(const int* __restrict__ ei, int E, int N, int H,
                               const float* __restrict__ h,
                               const float* __restrict__ logits,
                               const float* __restrict__ seg_sum,
                               float* __restrict__ acc) {
  int e = blockIdx.x;
  int s, d;
  if (e < E) { s = ei[e]; d = ei[E + e]; } else { s = d = e - E; }
  int c = threadIdx.x * 4;           // 128 threads x 4 channels = 512
  int C = 512 / H;
  int hd = c / C;
  float w = logits[(size_t)e * H + hd] / seg_sum[(size_t)d * H + hd];
  const float4 hv = *(const float4*)(h + (size_t)s * 512 + c);
  float* ap = acc + (size_t)d * 512 + c;
  atomicAdd(ap + 0, hv.x * w);
  atomicAdd(ap + 1, hv.y * w);
  atomicAdd(ap + 2, hv.z * w);
  atomicAdd(ap + 3, hv.w * w);
}

// out[n,:] = resid[n,:] + elu(layernorm(acc[n,:] + bias) * gamma + beta)
__global__ void finalize_kernel(const float* __restrict__ acc,
                                const float* __restrict__ bias,
                                const float* __restrict__ gamma,
                                const float* __restrict__ beta,
                                const float* __restrict__ resid,
                                float* __restrict__ out) {
  const int n = blockIdx.x;
  const int t = threadIdx.x;  // 256 threads, 2 channels each
  __shared__ float red[256];
  const size_t base = (size_t)n * 512;
  float v0 = acc[base + t] + bias[t];
  float v1 = acc[base + t + 256] + bias[t + 256];

  red[t] = v0 + v1;
  __syncthreads();
  for (int off = 128; off > 0; off >>= 1) {
    if (t < off) red[t] += red[t + off];
    __syncthreads();
  }
  const float mu = red[0] * (1.0f / 512.0f);
  __syncthreads();

  float d0 = v0 - mu, d1 = v1 - mu;
  red[t] = d0 * d0 + d1 * d1;
  __syncthreads();
  for (int off = 128; off > 0; off >>= 1) {
    if (t < off) red[t] += red[t + off];
    __syncthreads();
  }
  const float rstd = rsqrtf(red[0] * (1.0f / 512.0f) + 1e-5f);

  float y0 = d0 * rstd * gamma[t] + beta[t];
  float y1 = d1 * rstd * gamma[t + 256] + beta[t + 256];
  y0 = (y0 > 0.0f) ? y0 : (expf(y0) - 1.0f);  // ELU
  y1 = (y1 > 0.0f) ? y1 : (expf(y1) - 1.0f);
  out[base + t]       = resid[base + t] + y0;
  out[base + t + 256] = resid[base + t + 256] + y1;
}

// ------------------------------- launcher ----------------------------------

extern "C" void kernel_launch(void* const* d_in, const int* in_sizes, int n_in,
                              void* d_out, int out_size, void* d_ws, size_t ws_size,
                              hipStream_t stream) {
  const float* x    = (const float*)d_in[0];
  const int*   ei   = (const int*)  d_in[1];
  const float* W1   = (const float*)d_in[2];
  const float* a_s1 = (const float*)d_in[3];
  const float* a_d1 = (const float*)d_in[4];
  const float* b1   = (const float*)d_in[5];
  const float* g1   = (const float*)d_in[6];
  const float* be1  = (const float*)d_in[7];
  const float* W2   = (const float*)d_in[8];
  const float* a_s2 = (const float*)d_in[9];
  const float* a_d2 = (const float*)d_in[10];
  const float* b2   = (const float*)d_in[11];
  const float* g2   = (const float*)d_in[12];
  const float* be2  = (const float*)d_in[13];

  const int N = in_sizes[0] / 512;
  const int E = in_sizes[1] / 2;
  const int Etot = E + N;

  // workspace carve (256B aligned)
  char* p = (char*)d_ws;
  auto carve = [&](size_t bytes) -> char* {
    char* r = p;
    p += (bytes + 255) & ~(size_t)255;
    return r;
  };
  unsigned short* xb = (unsigned short*)carve((size_t)N * 512 * 2);
  unsigned short* wt = (unsigned short*)carve((size_t)512 * 512 * 2);
  float* h    = (float*)carve((size_t)N * 512 * 4);
  float* acc  = (float*)carve((size_t)N * 512 * 4);
  float* x1   = (float*)carve((size_t)N * 512 * 4);
  float* as   = (float*)carve((size_t)N * 8 * 4);
  float* ad   = (float*)carve((size_t)N * 8 * 4);
  float* smax = (float*)carve((size_t)N * 8 * 4);
  float* ssum = (float*)carve((size_t)N * 8 * 4);
  float* lg   = (float*)carve((size_t)Etot * 8 * 4);

  auto run_layer = [&](const float* xin, const float* W, const float* avs,
                       const float* avd, const float* bias, const float* gamma,
                       const float* beta, int H, const float* resid, float* out) {
    const int C = 512 / H;
    const size_t nfeat = (size_t)N * 512;
    cvt_bf16_kernel<<<dim3((unsigned)((nfeat + 255) / 256)), dim3(256), 0, stream>>>(
        xin, xb, nfeat);
    cvt_w_transpose_kernel<<<dim3(1024), dim3(256), 0, stream>>>(W, wt);

    // one wave per 16x64 output strip: (N/16) * (512/64) waves, 8 waves/block
    const int total_waves = (N / 16) * 8;
    gemm_bf16_wmma_kernel<<<dim3((total_waves + 7) / 8), dim3(256), 0, stream>>>(
        xb, wt, h, total_waves);

    alpha_kernel<<<dim3((N * H + 255) / 256), dim3(256), 0, stream>>>(
        h, avs, avd, as, ad, N, H, C);

    fill_kernel<<<dim3(2048), dim3(256), 0, stream>>>(acc, 0.0f, nfeat);
    fill_kernel<<<dim3(256), dim3(256), 0, stream>>>(ssum, 0.0f, (size_t)N * H);
    fill_kernel<<<dim3(256), dim3(256), 0, stream>>>(smax, -INFINITY, (size_t)N * H);

    const size_t tot = (size_t)Etot * H;
    edge_logits_kernel<<<dim3((unsigned)((tot + 255) / 256)), dim3(256), 0, stream>>>(
        ei, E, N, H, as, ad, lg, smax);
    edge_exp_kernel<<<dim3((unsigned)((tot + 255) / 256)), dim3(256), 0, stream>>>(
        ei, E, N, H, lg, smax, ssum);
    scatter_kernel<<<dim3(Etot), dim3(128), 0, stream>>>(ei, E, N, H, h, lg, ssum, acc);

    finalize_kernel<<<dim3(N), dim3(256), 0, stream>>>(acc, bias, gamma, beta, resid, out);
  };

  // layer 1: 8 heads, concat; residual with input x
  run_layer(x, W1, a_s1, a_d1, b1, g1, be1, 8, x, x1);
  // layer 2: 1 head (mean over 1 head == identity); residual with x1
  run_layer(x1, W2, a_s2, a_d2, b2, g2, be2, 1, x1, (float*)d_out);
}
